// SwinTransformerBlock_24507083391121
// MI455X (gfx1250) — matrix operand order repaired and checked by reading, not compiled, not run on hardware
//
#include <hip/hip_runtime.h>
#include <hip/hip_bf16.h>

typedef __attribute__((ext_vector_type(16))) __bf16 v16bf;
typedef __attribute__((ext_vector_type(8)))  __bf16 v8bf;
typedef __attribute__((ext_vector_type(8)))  float  v8f;
typedef __attribute__((ext_vector_type(4)))  int    v4i;

#define DI __device__ __forceinline__

#if defined(__has_builtin)
#if __has_builtin(__builtin_amdgcn_global_load_async_to_lds_b128)
#define USE_ASYNC_LDS 1
#endif
#endif

DI v16bf ld16x2(const __bf16* p0, const __bf16* p1) {
    v8bf a = *(const v8bf*)p0;
    v8bf b = *(const v8bf*)p1;
    return __builtin_shufflevector(a, b, 0,1,2,3,4,5,6,7,8,9,10,11,12,13,14,15);
}

DI v8f wmma_bf16(v16bf a, v16bf b, v8f c) {
    // v_wmma_f32_16x16x32_bf16 : D = A(16x32) * B(32x16) + C, f32 accum
    return __builtin_amdgcn_wmma_f32_16x16x32_bf16(false, a, false, b, (short)0, c, false, false);
}

// stage 16 bytes global -> LDS (async when available: global_load_async_to_lds_b128)
DI void stage_b128(const __bf16* gsrc, __bf16* ldst) {
#ifdef USE_ASYNC_LDS
    typedef __attribute__((address_space(1))) v4i* gp1_t;   // global int4*
    typedef __attribute__((address_space(3))) v4i* lp3_t;   // LDS int4*
    __builtin_amdgcn_global_load_async_to_lds_b128(
        (gp1_t)(__UINTPTR_TYPE__)gsrc,
        (lp3_t)(unsigned)(__UINTPTR_TYPE__)ldst, 0, 0);
#else
    *(v8bf*)ldst = *(const v8bf*)gsrc;
#endif
}

DI void stage_wait() {
#ifdef USE_ASYNC_LDS
    asm volatile("s_wait_asynccnt 0" ::: "memory");
#endif
}

// ---------------------------------------------------------------------------
// Problem constants (B=8, H=W=128, C=256, ws=8, shift=4, heads=8, d=32)
// tokens T = 131072, windows B_ = 2048, N = 64, hidden = 1024
// ---------------------------------------------------------------------------

// ---- weight fp32 -> bf16 conversion ---------------------------------------
__global__ __launch_bounds__(256) void cvt_weights(const float* __restrict__ qkv_w,
                                                   const float* __restrict__ proj_w,
                                                   const float* __restrict__ fc1_w,
                                                   const float* __restrict__ fc2_w,
                                                   __bf16* __restrict__ wbf) {
    int i = blockIdx.x * 256 + threadIdx.x;
    float v;
    if      (i < 196608) v = qkv_w[i];
    else if (i < 262144) v = proj_w[i - 196608];
    else if (i < 524288) v = fc1_w[i - 262144];
    else if (i < 786432) v = fc2_w[i - 524288];
    else return;
    wbf[i] = (__bf16)v;
}

// ---- LayerNorm (one wave per token, lane = 8 channels) --------------------
template<bool SHIFTED>
__global__ __launch_bounds__(256) void ln_kernel(const float* __restrict__ x,
                                                 const float* __restrict__ gamma,
                                                 const float* __restrict__ beta,
                                                 __bf16* __restrict__ out) {
    const int lane = threadIdx.x & 31;
    const int wave = threadIdx.x >> 5;
    const long t = (long)blockIdx.x * 8 + wave;   // destination token (window order)
    long src_tok;
    if (SHIFTED) {
        int b_ = (int)(t >> 6), n = (int)(t & 63);
        int b = b_ >> 8, wh = (b_ >> 4) & 15, ww = b_ & 15;
        int r = n >> 3, c = n & 7;
        int hp = ((wh << 3) + r + 4) & 127;       // roll(-shift)
        int wp = ((ww << 3) + c + 4) & 127;
        src_tok = (long)b * 16384 + hp * 128 + wp;
    } else {
        src_tok = t;
    }
    const float* src = x + src_tok * 256 + lane * 8;
    float4 v0 = *(const float4*)(src);
    float4 v1 = *(const float4*)(src + 4);
    float vals[8] = {v0.x, v0.y, v0.z, v0.w, v1.x, v1.y, v1.z, v1.w};
    float s = 0.f, q = 0.f;
#pragma unroll
    for (int i = 0; i < 8; ++i) { s += vals[i]; q += vals[i] * vals[i]; }
#pragma unroll
    for (int m = 1; m < 32; m <<= 1) {
        s += __shfl_xor(s, m, 32);
        q += __shfl_xor(q, m, 32);
    }
    float mean = s * (1.f / 256.f);
    float var  = q * (1.f / 256.f) - mean * mean;
    float rstd = rsqrtf(var + 1e-5f);
    const float* g = gamma + lane * 8;
    const float* b = beta + lane * 8;
    v8bf o = {};
#pragma unroll
    for (int i = 0; i < 8; ++i)
        o[i] = (__bf16)((vals[i] - mean) * rstd * g[i] + b[i]);
    *(v8bf*)(out + t * 256 + lane * 8) = o;
}

// ---- bf16 GEMM with block-cooperative LDS staging of the weight tile ------
// block = 8 waves x 32 rows = 256 rows, all sharing one 64-column tile.
// Per K-step: 64x32 bf16 B tile staged to LDS (async b128 when available),
// fragments read back with ds_load; A streamed straight from global.
// EPI: 0 = QKV scatter, 1 = proj+window-reverse+residual, 2 = FC1+GELU, 3 = FC2+residual
#define BSTRIDE 40   // padded LDS row stride (bank-conflict-free fragment reads)

template<int EPI>
__global__ __launch_bounds__(256) void gemm_bf16_kernel(
        const __bf16* __restrict__ A, const __bf16* __restrict__ W,
        const float* __restrict__ bias, const int K, const int Ntiles,
        float* __restrict__ outf, const float* __restrict__ resid,
        __bf16* __restrict__ outb) {
    __shared__ __bf16 Bt[64 * BSTRIDE];

    const int tid  = threadIdx.x;
    const int lane = tid & 31;
    const int wave = tid >> 5;
    const int half = lane >> 4;
    const int l15  = lane & 15;
    const int tn = blockIdx.x % Ntiles;
    const long rowbase = ((long)(blockIdx.x / Ntiles) * 8 + wave) * 32;
    const int colbase  = tn * 64;

    // staging assignment: thread -> (column c, 8-element segment)
    const int sc  = tid >> 2;        // 0..63
    const int seg = (tid & 3) * 8;   // 0,8,16,24
    const __bf16* wsrc = W + (size_t)(colbase + sc) * K + seg;
    __bf16* ldst = &Bt[sc * BSTRIDE + seg];

    v8f acc[2][4] = {};

    const __bf16* pa0 = A + (rowbase + l15) * K + 16 * half;
    const __bf16* pa1 = A + (rowbase + 16 + l15) * K + 16 * half;

    for (int k0 = 0; k0 < K; k0 += 32) {
        __syncthreads();                       // previous tile fully consumed
        stage_b128(wsrc + k0, ldst);           // 16B per thread, 512B per wave
        stage_wait();                          // s_wait_asynccnt 0
        __syncthreads();                       // tile visible to all waves

        v16bf af0 = ld16x2(pa0 + k0, pa0 + k0 + 8);
        v16bf af1 = ld16x2(pa1 + k0, pa1 + k0 + 8);
        __builtin_prefetch(pa0 + k0 + 32, 0, 1);   // global_prefetch_b8
#pragma unroll
        for (int nt = 0; nt < 4; ++nt) {
            const __bf16* pb = &Bt[(nt * 16 + l15) * BSTRIDE + 16 * half];
            v16bf bfrag = ld16x2(pb, pb + 8);      // ds_load_b128 x2
            acc[0][nt] = wmma_bf16(af0, bfrag, acc[0][nt]);
            acc[1][nt] = wmma_bf16(af1, bfrag, acc[1][nt]);
        }
    }

#pragma unroll
    for (int mt = 0; mt < 2; ++mt)
#pragma unroll
    for (int nt = 0; nt < 4; ++nt)
#pragma unroll
    for (int r = 0; r < 8; ++r) {
        const long row = rowbase + mt * 16 + r + 8 * half;
        const int col = colbase + nt * 16 + l15;
        float v = acc[mt][nt][r] + bias[col];
        if constexpr (EPI == 0) {
            // scatter into per-(window,head) q/k/v blocks of 64x32
            int which = col >> 8, head = (col >> 5) & 7, di = col & 31;
            int b_ = (int)(row >> 6), n = (int)(row & 63);
            outb[((size_t)which * 16384 + (b_ << 3) + head) * 2048 + (n << 5) + di] = (__bf16)v;
        } else if constexpr (EPI == 1) {
            // window-reverse + reverse cyclic shift + residual (x2 = x + attn)
            int b_ = (int)(row >> 6), n = (int)(row & 63);
            int b = b_ >> 8, wh = (b_ >> 4) & 15, ww = b_ & 15;
            int r2 = n >> 3, c2 = n & 7;
            int hp = ((wh << 3) + r2 + 4) & 127;
            int wp = ((ww << 3) + c2 + 4) & 127;
            size_t idx = ((size_t)b * 16384 + hp * 128 + wp) * 256 + col;
            outf[idx] = resid[idx] + v;
        } else if constexpr (EPI == 2) {
            float g = 0.5f * v * (1.f + erff(v * 0.70710678118654752f));  // exact GELU
            outb[(size_t)row * 1024 + col] = (__bf16)g;
        } else {
            size_t idx = (size_t)row * 256 + col;    // outf holds x2; out = x2 + mlp
            outf[idx] = outf[idx] + v;
        }
    }
}

// ---- fused window attention: one wave per (window, head) -------------------
// S = scale*Q.K^T + rel_bias + mask; P = softmax(S); O = P.V
__global__ __launch_bounds__(32) void attn_kernel(const __bf16* __restrict__ qkv,
                                                  const float* __restrict__ relb,
                                                  const float* __restrict__ mask,
                                                  __bf16* __restrict__ out) {
    __shared__ __bf16 P[64 * 64];
    const int lane = threadIdx.x;
    const int half = lane >> 4, l15 = lane & 15;
    const int bh = blockIdx.x;          // = b_ * 8 + head
    const int b_ = bh >> 3, head = bh & 7;
    const int w = b_ & 255;             // mask window index

    const __bf16* q  = qkv + (size_t)bh * 2048;
    const __bf16* kk = qkv + ((size_t)16384 + bh) * 2048;
    const __bf16* vv = qkv + ((size_t)32768 + bh) * 2048;

    // Q as A-fragments (single K-step, d = 32)
    v16bf aq[4];
#pragma unroll
    for (int mt = 0; mt < 4; ++mt) {
        const __bf16* p = q + (mt * 16 + l15) * 32 + 16 * half;
        aq[mt] = ld16x2(p, p + 8);
    }
    // K as B-fragments (column n = key token, contiguous d)
    v16bf bk[4];
#pragma unroll
    for (int nt = 0; nt < 4; ++nt) {
        const __bf16* p = kk + (nt * 16 + l15) * 32 + 16 * half;
        bk[nt] = ld16x2(p, p + 8);
    }

    v8f s[4][4] = {};
#pragma unroll
    for (int mt = 0; mt < 4; ++mt)
#pragma unroll
    for (int nt = 0; nt < 4; ++nt)
        s[mt][nt] = wmma_bf16(aq[mt], bk[nt], s[mt][nt]);

    const float scale = 0.17677669529663687f;   // 1/sqrt(32)
    const float* mrow = mask + (size_t)w * 4096;
#pragma unroll
    for (int mt = 0; mt < 4; ++mt)
#pragma unroll
    for (int r = 0; r < 8; ++r) {
        const int m = mt * 16 + r + 8 * half;
        const int mi = m >> 3, mj = m & 7;
        float mx = -3.0e38f;
#pragma unroll
        for (int nt = 0; nt < 4; ++nt) {
            const int n = nt * 16 + l15;
            const int ri = mi - (n >> 3) + 7;
            const int ci = mj - (n & 7) + 7;
            float val = s[mt][nt][r] * scale + relb[(ri * 15 + ci) * 8 + head]
                      + mrow[m * 64 + n];
            s[mt][nt][r] = val;
            mx = fmaxf(mx, val);
        }
        mx = fmaxf(mx, __shfl_xor(mx, 1, 32));
        mx = fmaxf(mx, __shfl_xor(mx, 2, 32));
        mx = fmaxf(mx, __shfl_xor(mx, 4, 32));
        mx = fmaxf(mx, __shfl_xor(mx, 8, 32));
        float sm = 0.f;
#pragma unroll
        for (int nt = 0; nt < 4; ++nt) {
            float e = __expf(s[mt][nt][r] - mx);
            s[mt][nt][r] = e;
            sm += e;
        }
        sm += __shfl_xor(sm, 1, 32);
        sm += __shfl_xor(sm, 2, 32);
        sm += __shfl_xor(sm, 4, 32);
        sm += __shfl_xor(sm, 8, 32);
        const float inv = 1.f / sm;
#pragma unroll
        for (int nt = 0; nt < 4; ++nt)
            P[m * 64 + nt * 16 + l15] = (__bf16)(s[mt][nt][r] * inv);
    }

    asm volatile("s_wait_dscnt 0" ::: "memory");   // P stores visible to ds loads

    v8f o[4][2] = {};
#pragma unroll
    for (int ks = 0; ks < 2; ++ks) {
        v16bf ap[4];
#pragma unroll
        for (int mt = 0; mt < 4; ++mt) {
            const __bf16* pp = &P[(mt * 16 + l15) * 64 + ks * 32 + 16 * half];
            ap[mt] = ld16x2(pp, pp + 8);
        }
        v16bf bv[2];
#pragma unroll
        for (int nt = 0; nt < 2; ++nt) {
            const int nd = nt * 16 + l15;
            v16bf t = {};
#pragma unroll
            for (int e = 0; e < 16; ++e)
                t[e] = vv[(ks * 32 + 16 * half + e) * 32 + nd];
            bv[nt] = t;
        }
#pragma unroll
        for (int mt = 0; mt < 4; ++mt)
#pragma unroll
        for (int nt = 0; nt < 2; ++nt)
            o[mt][nt] = wmma_bf16(ap[mt], bv[nt], o[mt][nt]);
    }

    __bf16* ob = out + (size_t)b_ * 64 * 256 + head * 32;
#pragma unroll
    for (int mt = 0; mt < 4; ++mt)
#pragma unroll
    for (int nt = 0; nt < 2; ++nt)
#pragma unroll
    for (int r = 0; r < 8; ++r) {
        const int m = mt * 16 + r + 8 * half;
        const int nd = nt * 16 + l15;
        ob[(size_t)m * 256 + nd] = (__bf16)o[mt][nt][r];
    }
}

// ---------------------------------------------------------------------------
extern "C" void kernel_launch(void* const* d_in, const int* in_sizes, int n_in,
                              void* d_out, int out_size, void* d_ws, size_t ws_size,
                              hipStream_t stream) {
    const float* x      = (const float*)d_in[0];
    const float* maskm  = (const float*)d_in[1];
    const float* n1g    = (const float*)d_in[2];
    const float* n1b    = (const float*)d_in[3];
    const float* qkv_w  = (const float*)d_in[4];
    const float* qkv_b  = (const float*)d_in[5];
    const float* relb   = (const float*)d_in[6];
    const float* proj_w = (const float*)d_in[7];
    const float* proj_b = (const float*)d_in[8];
    const float* n2g    = (const float*)d_in[9];
    const float* n2b    = (const float*)d_in[10];
    const float* fc1_w  = (const float*)d_in[11];
    const float* fc1_b  = (const float*)d_in[12];
    const float* fc2_w  = (const float*)d_in[13];
    const float* fc2_b  = (const float*)d_in[14];
    float* out = (float*)d_out;

    char* ws = (char*)d_ws;
    // region A (256 MB): qkv bf16 (3*2048*8 blocks of 64x32), later FC1 output h
    __bf16* regA = (__bf16*)(ws);
    // region B (64 MB): xn -> attnout -> y_in (all 131072 x 256 bf16)
    __bf16* regB = (__bf16*)(ws + 268435456);
    // bf16 weights (~1.5 MB)
    __bf16* wbf  = (__bf16*)(ws + 268435456 + 67108864);
    __bf16* wqkv = wbf;
    __bf16* wproj = wbf + 196608;
    __bf16* wfc1 = wbf + 262144;
    __bf16* wfc2 = wbf + 524288;

    cvt_weights<<<3072, 256, 0, stream>>>(qkv_w, proj_w, fc1_w, fc2_w, wbf);

    // LN1 + cyclic shift + window partition -> bf16 (131072, 256)
    ln_kernel<true><<<16384, 256, 0, stream>>>(x, n1g, n1b, regB);

    // QKV: (131072,256) x (768,256)^T, scatter to q/k/v per (window,head)
    gemm_bf16_kernel<0><<<6144, 256, 0, stream>>>(regB, wqkv, qkv_b, 256, 12,
                                                  nullptr, nullptr, regA);

    // fused window attention -> attnout (131072, 256) bf16
    attn_kernel<<<16384, 32, 0, stream>>>(regA, relb, maskm, regB);

    // proj + window reverse + reverse shift + residual -> d_out = x2 (fp32)
    gemm_bf16_kernel<1><<<2048, 256, 0, stream>>>(regB, wproj, proj_b, 256, 4,
                                                  out, x, nullptr);

    // LN2(x2) -> bf16
    ln_kernel<false><<<16384, 256, 0, stream>>>(out, n2g, n2b, regB);

    // FC1 + exact GELU -> h (131072, 1024) bf16
    gemm_bf16_kernel<2><<<8192, 256, 0, stream>>>(regB, wfc1, fc1_b, 256, 16,
                                                  nullptr, nullptr, regA);

    // FC2 + residual (in place on d_out = x2)
    gemm_bf16_kernel<3><<<2048, 256, 0, stream>>>(regA, wfc2, fc2_b, 1024, 4,
                                                  out, nullptr, nullptr);
}
